// TemplatePointwiseAttention_26242250179321
// MI455X (gfx1250) — matrix operand order repaired and checked by reading, not compiled, not run on hardware
//
#include <hip/hip_runtime.h>
#include <hip/hip_bf16.h>

// ---- types ------------------------------------------------------------
typedef __bf16 bf16_t;
typedef bf16_t v16bf __attribute__((ext_vector_type(16)));
typedef bf16_t v8bf  __attribute__((ext_vector_type(8)));
typedef float  v8f   __attribute__((ext_vector_type(8)));

// ---- problem constants ------------------------------------------------
#define HC   256           // H*C
#define CZ   128           // pair-rep channels
#define CT   64            // template channels
#define TT   4             // templates
#define RR   384
#define PP   (RR*RR)       // 147456 residue pairs
#define WAVES 8
#define TILE  16           // pairs per wave

// ---- LDS layout (bytes) ----------------------------------------------
#define SM_MT   0          // MT: [256][128] bf16 = 65536  (B for U = Z @ Mcat)
#define SM_NT   65536      // NT: [128][256] bf16 = 65536  (B for O = S @ Ncat)
#define SM_TK   131072     // per wave: 16*4*64 bf16 = 8192  (x8 = 65536)
#define SM_U    196608     // per wave: 16*256  bf16 = 8192  (x8 = 65536)
#define SM_W    262144     // per wave: 16*4*4  f32  = 1024  (x8 = 8192)
#define SM_TOTAL 270336

// =======================================================================
// Precompute folded weights into d_ws (bf16):
//   MT[n][k]  , n=h*64+k2 : (1/8) * sum_c Wq[k, h*64+c] * Wk[k2, h*64+c]
//   NT[n2][m] , m=h*64+k2 :        sum_c Wv[k2, h*64+c] * Wo[h*64+c, n2]
// =======================================================================
__global__ __launch_bounds__(256) void tpa_precompute(
    const float* __restrict__ Wq, const float* __restrict__ Wk,
    const float* __restrict__ Wv, const float* __restrict__ Wo,
    bf16_t* __restrict__ Wout)
{
    int idx = blockIdx.x * 256 + threadIdx.x;   // 0 .. 65535
    if (idx < 32768) {                          // MT element
        int nn = idx >> 7, k = idx & 127;
        int h = nn >> 6, k2 = nn & 63;
        const float* wq = Wq + (size_t)k  * HC + h * 64;
        const float* wk = Wk + (size_t)k2 * HC + h * 64;
        float s = 0.f;
        for (int c = 0; c < 64; ++c) s = fmaf(wq[c], wk[c], s);
        Wout[idx] = (bf16_t)(0.125f * s);       // fold C^-0.5
    } else {                                    // NT element
        int e  = idx - 32768;
        int n2 = e >> 8, m = e & 255;
        int h = m >> 6, k2 = m & 63;
        const float* wv = Wv + (size_t)k2 * HC + h * 64;
        float s = 0.f;
        for (int c = 0; c < 64; ++c) s = fmaf(wv[c], Wo[(size_t)(h*64 + c) * CZ + n2], s);
        Wout[32768 + e] = (bf16_t)s;
    }
}

// =======================================================================
// Main fused kernel: one wave = 16 pairs.
//   U = Zb[16x128] @ Mcat[128x256]        (4 ksteps x 16 ntiles WMMA)
//   logits[p,h,t] = diag( U_h[16x64] @ Tk_t^T[64x16] )   (32 WMMA)
//   softmax over t -> w
//   S[p,h*64+c] = sum_t w[p,h,t]*Tk[p,t,c]  (built in A-layout)
//   O = S[16x256] @ Ncat[256x128]         (8 ksteps x 8 ntiles WMMA)
// =======================================================================
__global__ __launch_bounds__(256, 1) void tpa_main(
    const float* __restrict__ z2d, const float* __restrict__ t2d,
    const bf16_t* __restrict__ Wws, const float* __restrict__ bo,
    float* __restrict__ out)
{
    extern __shared__ char smem[];
    bf16_t* smMT = (bf16_t*)(smem + SM_MT);
    bf16_t* smNT = (bf16_t*)(smem + SM_NT);

    const int tid  = threadIdx.x;
    const int wave = tid >> 5;
    const int lane = tid & 31;
    const int n    = lane & 15;   // N-col / pair-row selector
    const int lh   = lane >> 4;   // lane half (wave32 WMMA layout)

    bf16_t* smTk = (bf16_t*)(smem + SM_TK) + wave * (TILE*TT*CT);
    bf16_t* smU  = (bf16_t*)(smem + SM_U)  + wave * (TILE*HC);
    float*  smL  = (float*) (smem + SM_W)  + wave * (TILE*TT*4); // logits, then weights

    // ---- async staging of folded weights into LDS (131072 B) ---------
    // GVS mode: global = SADDR + VADDR + offset ; LDS = VDST + offset.
    // Same offset advances both sides; layouts are identical.
    {
        unsigned ldsBase = (unsigned)(size_t)smem + (unsigned)tid * 16u;
        unsigned gOff    = (unsigned)tid * 16u;
        #pragma unroll
        for (int i = 0; i < 32; ++i) {
            asm volatile("global_load_async_to_lds_b128 %0, %1, %2 offset:%3"
                         :: "v"(ldsBase), "v"(gOff), "s"(Wws), "i"(i * 4096)
                         : "memory");
        }
        asm volatile("s_wait_asynccnt 0" ::: "memory");
    }
    __syncthreads();

    const int tile = blockIdx.x * WAVES + wave;
    const int p0   = tile * TILE;

    // ---- stage Tk tile into LDS (fp32 global -> bf16 LDS) -------------
    {
        const int pl = lane >> 1;          // pair handled by this lane
        const int cb = (lane & 1) * 32;    // channel half
        #pragma unroll
        for (int t = 0; t < TT; ++t) {
            const float* src = t2d + ((size_t)t * PP + (size_t)(p0 + pl)) * CT + cb;
            bf16_t* dst = smTk + (pl*TT + t)*CT + cb;
            #pragma unroll
            for (int q = 0; q < 4; ++q) {
                float4 f0 = ((const float4*)src)[2*q];
                float4 f1 = ((const float4*)src)[2*q + 1];
                v8bf v;
                v[0]=(bf16_t)f0.x; v[1]=(bf16_t)f0.y; v[2]=(bf16_t)f0.z; v[3]=(bf16_t)f0.w;
                v[4]=(bf16_t)f1.x; v[5]=(bf16_t)f1.y; v[6]=(bf16_t)f1.z; v[7]=(bf16_t)f1.w;
                *(v8bf*)(dst + 8*q) = v;
            }
        }
    }

    const v8f vzero = {0.f,0.f,0.f,0.f,0.f,0.f,0.f,0.f};

    // ---- Stage A: U = Z @ Mcat ---------------------------------------
    v8f accU[16];
    #pragma unroll
    for (int j = 0; j < 16; ++j) accU[j] = vzero;

    const float* zrow = z2d + (size_t)(p0 + n) * CZ;   // A row = pair n
    #pragma unroll
    for (int kk = 0; kk < 4; ++kk) {
        // A-fragment: elems 0..15 = K 32kk+16lh+{0..15}
        const float* c0 = zrow + 32*kk + 16*lh;
        float4 a0 = ((const float4*)c0)[0];
        float4 a1 = ((const float4*)c0)[1];
        float4 b0 = ((const float4*)(c0 + 8))[0];
        float4 b1 = ((const float4*)(c0 + 8))[1];
        v16bf A;
        A[0]=(bf16_t)a0.x;  A[1]=(bf16_t)a0.y;  A[2]=(bf16_t)a0.z;  A[3]=(bf16_t)a0.w;
        A[4]=(bf16_t)a1.x;  A[5]=(bf16_t)a1.y;  A[6]=(bf16_t)a1.z;  A[7]=(bf16_t)a1.w;
        A[8]=(bf16_t)b0.x;  A[9]=(bf16_t)b0.y;  A[10]=(bf16_t)b0.z; A[11]=(bf16_t)b0.w;
        A[12]=(bf16_t)b1.x; A[13]=(bf16_t)b1.y; A[14]=(bf16_t)b1.z; A[15]=(bf16_t)b1.w;
        #pragma unroll
        for (int j = 0; j < 16; ++j) {
            // B-fragment: lane holds col N=16j+n, K = 32kk + 16lh + e (contiguous 32B)
            v16bf B = *(const v16bf*)(smMT + (size_t)(16*j + n)*CZ + 32*kk + 16*lh);
            accU[j] = __builtin_amdgcn_wmma_f32_16x16x32_bf16(
                false, A, false, B, (short)0, accU[j], false, false);
        }
    }

    // ---- spill U to LDS (bf16), row-major [pair][HC] -------------------
    #pragma unroll
    for (int j = 0; j < 16; ++j) {
        #pragma unroll
        for (int r = 0; r < 8; ++r)
            smU[(size_t)(r + 8*lh)*HC + 16*j + n] = (bf16_t)accU[j][r];
    }

    // ---- logits via WMMA diagonal -------------------------------------
    // D = U_h[16x64] @ Tk_t^T[64x16]; logits[p,h,t] = D[p][p].
    {
        #pragma unroll
        for (int h = 0; h < 4; ++h) {
            const bf16_t* urow = smU + (size_t)n*HC + h*64;  // A row = pair n
            v8bf ua0 = *(const v8bf*)(urow + 16*lh);
            v8bf ua1 = *(const v8bf*)(urow + 8 + 16*lh);
            v8bf ua2 = *(const v8bf*)(urow + 32 + 16*lh);
            v8bf ua3 = *(const v8bf*)(urow + 40 + 16*lh);
            v16bf A0 = __builtin_shufflevector(ua0, ua1, 0,1,2,3,4,5,6,7,8,9,10,11,12,13,14,15);
            v16bf A1 = __builtin_shufflevector(ua2, ua3, 0,1,2,3,4,5,6,7,8,9,10,11,12,13,14,15);
            #pragma unroll
            for (int t = 0; t < 4; ++t) {
                // B-fragment: col N = pair n, K = 32ks + 16lh + e (contiguous 32B)
                const bf16_t* tk = smTk + (n*TT + t)*CT;
                v16bf B0 = *(const v16bf*)(tk + 16*lh);
                v16bf B1 = *(const v16bf*)(tk + 32 + 16*lh);
                v8f D = vzero;
                D = __builtin_amdgcn_wmma_f32_16x16x32_bf16(false, A0, false, B0, (short)0, D, false, false);
                D = __builtin_amdgcn_wmma_f32_16x16x32_bf16(false, A1, false, B1, (short)0, D, false, false);
                // diagonal element for col n lives at row r = n - 8*lh in this lane
                float dv = D[0];
                #pragma unroll
                for (int r = 1; r < 8; ++r) dv = ((n - 8*lh) == r) ? D[r] : dv;
                if ((n >> 3) == lh) smL[(n*4 + h)*4 + t] = dv;
            }
        }
    }

    // ---- softmax: lane -> (pair n, heads 2lh..2lh+1); w overwrites L --
    {
        #pragma unroll
        for (int hh = 0; hh < 2; ++hh) {
            const int h = 2*lh + hh;
            float4 Lv = *(const float4*)(smL + (n*4 + h)*4);
            float m  = fmaxf(fmaxf(Lv.x, Lv.y), fmaxf(Lv.z, Lv.w));
            float e0 = __expf(Lv.x - m), e1 = __expf(Lv.y - m);
            float e2 = __expf(Lv.z - m), e3 = __expf(Lv.w - m);
            float inv = 1.f / (e0 + e1 + e2 + e3);
            float4 wv = make_float4(e0*inv, e1*inv, e2*inv, e3*inv);
            *(float4*)(smL + (n*4 + h)*4) = wv;
        }
    }

    // ---- S (weighted template sum) in A-layout, then O = S @ Ncat -----
    v8f accO[8];
    #pragma unroll
    for (int j = 0; j < 8; ++j) accO[j] = vzero;
    {
        // hoist this lane's Tk chunks to fp32 once: c = 32*(ch2>>1) + 16*lh + 8*(ch2&1) + i
        float tkf[4][4][8];
        #pragma unroll
        for (int t = 0; t < 4; ++t) {
            const bf16_t* tk = smTk + (n*TT + t)*CT;
            #pragma unroll
            for (int ch2 = 0; ch2 < 4; ++ch2) {
                v8bf tv = *(const v8bf*)(tk + 32*(ch2>>1) + 8*(ch2&1) + 16*lh);
                #pragma unroll
                for (int i = 0; i < 8; ++i) tkf[t][ch2][i] = (float)tv[i];
            }
        }
        #pragma unroll
        for (int kk = 0; kk < 8; ++kk) {
            const int h = kk >> 1;                       // K=64h+c, h const per kstep
            float4 wv = *(const float4*)(smL + (n*4 + h)*4);
            const int c0 = (kk & 1) * 2;                 // chunk index for elems 0..7
            const int c1 = c0 + 1;                       // chunk index for elems 8..15
            v16bf A;
            #pragma unroll
            for (int i = 0; i < 8; ++i) {
                float s0 = fmaf(wv.x, tkf[0][c0][i],
                           fmaf(wv.y, tkf[1][c0][i],
                           fmaf(wv.z, tkf[2][c0][i], wv.w * tkf[3][c0][i])));
                float s1 = fmaf(wv.x, tkf[0][c1][i],
                           fmaf(wv.y, tkf[1][c1][i],
                           fmaf(wv.z, tkf[2][c1][i], wv.w * tkf[3][c1][i])));
                A[i]     = (bf16_t)s0;
                A[8 + i] = (bf16_t)s1;
            }
            #pragma unroll
            for (int j = 0; j < 8; ++j) {
                v16bf B = *(const v16bf*)(smNT + (size_t)(16*j + n)*HC + 32*kk + 16*lh);
                accO[j] = __builtin_amdgcn_wmma_f32_16x16x32_bf16(
                    false, A, false, B, (short)0, accO[j], false, false);
            }
        }
    }

    // ---- write output (+ bias) ----------------------------------------
    #pragma unroll
    for (int j = 0; j < 8; ++j) {
        float bv = bo[16*j + n];
        #pragma unroll
        for (int r = 0; r < 8; ++r) {
            out[(size_t)(p0 + r + 8*lh)*CZ + 16*j + n] = accO[j][r] + bv;
        }
    }
}

// =======================================================================
extern "C" void kernel_launch(void* const* d_in, const int* in_sizes, int n_in,
                              void* d_out, int out_size, void* d_ws, size_t ws_size,
                              hipStream_t stream) {
    const float* z2d = (const float*)d_in[0];
    const float* t2d = (const float*)d_in[1];
    const float* Wq  = (const float*)d_in[2];
    const float* Wk  = (const float*)d_in[3];
    const float* Wv  = (const float*)d_in[4];
    const float* Wo  = (const float*)d_in[5];
    const float* bo  = (const float*)d_in[6];
    float*  out = (float*)d_out;
    bf16_t* W   = (bf16_t*)d_ws;          // 131072 bytes: MT then NT

    (void)in_sizes; (void)n_in; (void)out_size; (void)ws_size;

    // fold weights (tiny): 65536 elements
    tpa_precompute<<<256, 256, 0, stream>>>(Wq, Wk, Wv, Wo, W);

    static_assert(SM_TOTAL <= 320 * 1024, "LDS budget");
    hipFuncSetAttribute(reinterpret_cast<const void*>(tpa_main),
                        hipFuncAttributeMaxDynamicSharedMemorySize, SM_TOTAL);

    // 147456 pairs / (8 waves * 16 pairs) = 1152 workgroups
    tpa_main<<<PP / (WAVES * TILE), 256, SM_TOTAL, stream>>>(z2d, t2d, W, bo, out);
}
